// GraphormerNet_9491877724300
// MI455X (gfx1250) — compile-verified, matrix operand order checked by reading, not run on hardware
//
#include <hip/hip_runtime.h>
#include <hip/hip_bf16.h>

typedef __attribute__((ext_vector_type(16))) _Float16 v16h;
typedef __attribute__((ext_vector_type(8)))  _Float16 v8h;
typedef __attribute__((ext_vector_type(8)))  float    v8f;
typedef __attribute__((ext_vector_type(4)))  int      v4i;

#define WMMA_F16(a,b,c) __builtin_amdgcn_wmma_f32_16x16x32_f16(false,(a),false,(b),(short)0,(c),false,false)

// -------- gfx1250 async global->LDS path (guarded; sync fallback) ----------
#if defined(__has_builtin)
#if __has_builtin(__builtin_amdgcn_global_load_async_to_lds_b128)
#define HAVE_ASYNC_LDS 1
#endif
#endif

#ifdef HAVE_ASYNC_LDS
#if __has_builtin(__builtin_amdgcn_s_wait_asynccnt)
#define ASYNC_WAIT(n) __builtin_amdgcn_s_wait_asynccnt(n)
#else
#define ASYNC_WAIT(n) asm volatile("s_wait_asynccnt " #n)
#endif
#else
#define ASYNC_WAIT(n)
#endif

// ---------------------------------------------------------------------------
// Fragment helpers (wave32, 16x16x32 f16 WMMA)
// A-frag: lane L holds row M=L&15; halves 0..7 = K {0..7 | 8..15}, 8..15 = K {16..23 | 24..31}
// B-frag (B = W^T with W [N,K] row-major): lane L holds W row n, same K pattern.
// ---------------------------------------------------------------------------
__device__ __forceinline__ v16h frag_pack(v8h lo, v8h hi) {
  v16h r;
#pragma unroll
  for (int i = 0; i < 8; ++i) { r[i] = lo[i]; r[i + 8] = hi[i]; }
  return r;
}

__device__ __forceinline__ v16h load_frag(const _Float16* __restrict__ base,
                                          int row, int ld, int k0, int lane) {
  const _Float16* p = base + (size_t)row * ld + k0 + ((lane & 16) ? 8 : 0);
  return frag_pack(*(const v8h*)p, *(const v8h*)(p + 16));
}

// K=16 valid, zero-pad K 16..31 (for hd=16 attention QK^T)
__device__ __forceinline__ v16h load_frag_pad16(const _Float16* __restrict__ p) {
  v8h z;
#pragma unroll
  for (int i = 0; i < 8; ++i) z[i] = (_Float16)0.f;
  return frag_pack(*(const v8h*)p, z);
}

// Cooperative stage of W tile chunk [n0..n0+63][kc..kc+63] into LDS (64x64 halves)
__device__ __forceinline__ void stage_w(const _Float16* __restrict__ W, int ldw,
                                        int n0, int kc, _Float16* dst, int tid) {
#pragma unroll
  for (int i = 0; i < 4; ++i) {
    const int v    = tid + i * 128;       // 512 16B-vectors total, 128 threads
    const int row  = v >> 3;
    const int colh = (v & 7) * 8;
    const _Float16* src = W + (size_t)(n0 + row) * ldw + kc + colh;
    _Float16* d = dst + row * 64 + colh;
#ifdef HAVE_ASYNC_LDS
    __builtin_amdgcn_global_load_async_to_lds_b128(
        (__attribute__((address_space(1))) v4i*)src,
        (__attribute__((address_space(3))) v4i*)d, 0, 0);
#else
    *(v8h*)d = *(const v8h*)src;
#endif
  }
}

// ---------------------------------------------------------------------------
// WMMA GEMM: C[M,N] = A[M,K] @ W[N,K]^T + bias
// block = 128 threads (4 waves), wave tile = 16 rows x 64 cols, grid = (M/64, N/64)
// Weight tile double-buffered in LDS via async global->LDS copies.
// mode: 0=f32 out, 1=f16 out, 2=f16 out + relu, 3=f16 out in transposed-V layout
// ---------------------------------------------------------------------------
__global__ void wmma_gemm_kernel(const _Float16* __restrict__ A, int lda,
                                 const _Float16* __restrict__ W, int ldw,
                                 const float* __restrict__ bias,
                                 float* __restrict__ Cf, _Float16* __restrict__ Ch,
                                 int K, int ldc, int mode) {
  __shared__ _Float16 sW[2][64 * 64];   // 2 x 8KB
  const int tid  = threadIdx.x;
  const int lane = tid & 31;
  const int wv   = tid >> 5;
  const int m0   = blockIdx.x * 64 + wv * 16;
  const int n0   = blockIdx.y * 64;
  v8f acc[4];
#pragma unroll
  for (int t = 0; t < 4; ++t)
#pragma unroll
    for (int r = 0; r < 8; ++r) acc[t][r] = 0.f;

  const int arow = m0 + (lane & 15);
  const int koff = (lane & 16) ? 8 : 0;

  stage_w(W, ldw, n0, 0, sW[0], tid);
  int cc = 0;
  for (int kc = 0; kc < K; kc += 64, ++cc) {
    const _Float16* cur = sW[cc & 1];
    if (kc + 64 < K) {
      stage_w(W, ldw, n0, kc + 64, sW[(cc & 1) ^ 1], tid);   // prefetch next chunk
      __builtin_prefetch(A + (size_t)arow * lda + kc + 64, 0, 0);
      ASYNC_WAIT(4);    // current chunk's 4 async ops complete (in-order), next 4 in flight
    } else {
      ASYNC_WAIT(0);
    }
    __syncthreads();
#pragma unroll
    for (int kk = 0; kk < 64; kk += 32) {
      v16h fa = load_frag(A, arow, lda, kc + kk, lane);
#pragma unroll
      for (int t = 0; t < 4; ++t) {
        const _Float16* p = cur + (size_t)(t * 16 + (lane & 15)) * 64 + kk + koff;
        v16h fb = frag_pack(*(const v8h*)p, *(const v8h*)(p + 16));
        acc[t] = WMMA_F16(fa, fb, acc[t]);
      }
    }
    __syncthreads();    // protect buffer overwritten by next iteration's stage
  }

  const int mbase = m0 + ((lane >> 4) << 3);
#pragma unroll
  for (int t = 0; t < 4; ++t) {
    const int n = n0 + t * 16 + (lane & 15);
    const float bb = bias ? bias[n] : 0.f;
#pragma unroll
    for (int r = 0; r < 8; ++r) {
      const int m = mbase + r;
      float v = acc[t][r] + bb;
      if (mode == 0) {
        Cf[(size_t)m * ldc + n] = v;
      } else if (mode == 3) {
        // V stored transposed per (graph, head): Vt[((b*8+h)*16+d)*128 + i]
        const int b = m >> 7, i = m & 127, h = n >> 4, d = n & 15;
        Ch[(((size_t)(b * 8 + h)) * 16 + d) * 128 + i] = (_Float16)v;
      } else {
        if (mode == 2) v = fmaxf(v, 0.f);
        Ch[(size_t)m * ldc + n] = (_Float16)v;
      }
    }
  }
}

// ---------------------------------------------------------------------------
// Fused attention: one block per (graph, head). Np=128, hd=16.
// QK^T (WMMA, K padded to 32) -> LDS scores (64KB) -> bias+softmax in place,
// re-stored as f16 -> PV (WMMA, K=128, A from LDS, B from transposed V).
// ---------------------------------------------------------------------------
__global__ void attn_kernel(const _Float16* __restrict__ Q,
                            const _Float16* __restrict__ Kh,
                            const _Float16* __restrict__ Vt,
                            const int* __restrict__ dist,       // [256,127,127]
                            const float* __restrict__ sp,       // [10,8] (layer slice)
                            const float* __restrict__ ew,       // [8]
                            const float* __restrict__ eb,       // [8]
                            _Float16* __restrict__ AO) {
  __shared__ float sS[128 * 128];   // 64 KB
  const int b    = blockIdx.x >> 3;
  const int hh   = blockIdx.x & 7;
  const int lane = threadIdx.x & 31;
  const int wv   = threadIdx.x >> 5;       // 8 waves, 16 rows each
  const int m0   = wv * 16;
  const int qoff = (lane & 16) ? 8 : 0;

  // ---- phase 1: scores = (Q K^T) * 1/sqrt(hd) ----
  const _Float16* Qb = Q  + (size_t)b * 128 * 128 + hh * 16;
  const _Float16* Kb = Kh + (size_t)b * 128 * 128 + hh * 16;
  v16h fq = load_frag_pad16(Qb + (size_t)(m0 + (lane & 15)) * 128 + qoff);
  const int mbase = m0 + ((lane >> 4) << 3);
#pragma unroll
  for (int t = 0; t < 8; ++t) {
    v16h fk = load_frag_pad16(Kb + (size_t)(t * 16 + (lane & 15)) * 128 + qoff);
    v8f acc;
#pragma unroll
    for (int r = 0; r < 8; ++r) acc[r] = 0.f;
    acc = WMMA_F16(fq, fk, acc);
    const int n = t * 16 + (lane & 15);
#pragma unroll
    for (int r = 0; r < 8; ++r) sS[(size_t)(mbase + r) * 128 + n] = acc[r] * 0.25f;
  }
  __syncthreads();

  // ---- phase 2: structural bias + softmax, re-store probabilities as f16 ----
  const int row = threadIdx.x;
  if (row < 128) {
    const float ew0 = ew[hh], eb0 = eb[hh];
    float* s = sS + (size_t)row * 128;
    float mx = -1e30f;
    for (int j = 0; j < 128; ++j) {
      int dr = (row > 0 && j > 0)
                 ? dist[(size_t)b * 16129 + (row - 1) * 127 + (j - 1)] : 0;
      int dc = dr > 9 ? 9 : dr;
      float bias;
      if (dc == 9 && dr > 0) bias = -1e30f;                 // mask (far nodes)
      else bias = sp[dc * 8 + hh] + eb0 + ((dr > 0) ? ew0 : 0.f);
      float v = s[j] + bias;
      s[j] = v;
      mx = fmaxf(mx, v);
    }
    float sum = 0.f;
    for (int j = 0; j < 128; ++j) { float e = __expf(s[j] - mx); s[j] = e; sum += e; }
    const float inv = 1.f / sum;
    _Float16* ph = reinterpret_cast<_Float16*>(s);
    for (int j = 0; j < 128; ++j) ph[j] = (_Float16)(s[j] * inv);  // safe in-place (j/2 <= j)
  }
  __syncthreads();

  // ---- phase 3: AO = P @ V  (K = 128) ----
  const _Float16* P  = reinterpret_cast<const _Float16*>(sS);  // row stride 256 halves
  const _Float16* Vb = Vt + (size_t)(b * 8 + hh) * 16 * 128;
  v8f acc;
#pragma unroll
  for (int r = 0; r < 8; ++r) acc[r] = 0.f;
  const int prow = m0 + (lane & 15);
#pragma unroll
  for (int k0 = 0; k0 < 128; k0 += 32) {
    const _Float16* pp = P + (size_t)prow * 256 + k0 + qoff;
    v16h fa = frag_pack(*(const v8h*)pp, *(const v8h*)(pp + 16));
    v16h fb = load_frag(Vb, lane & 15, 128, k0, lane);
    acc = WMMA_F16(fa, fb, acc);
  }
  _Float16* AOb = AO + (size_t)b * 128 * 128 + hh * 16;
#pragma unroll
  for (int r = 0; r < 8; ++r)
    AOb[(size_t)(mbase + r) * 128 + (lane & 15)] = (_Float16)acc[r];
}

// ---------------------------------------------------------------------------
// Residual add + LayerNorm over D=128; one wave per row; writes f32 + f16.
// ---------------------------------------------------------------------------
__global__ void add_ln_kernel(float* __restrict__ h, _Float16* __restrict__ hH,
                              const float* __restrict__ delta,
                              const float* __restrict__ s, const float* __restrict__ bb) {
  const int row = blockIdx.x, lane = threadIdx.x;   // 32 threads
  float v[4], sum = 0.f;
  float* hr = h + (size_t)row * 128;
  const float* dr = delta + (size_t)row * 128;
#pragma unroll
  for (int e = 0; e < 4; ++e) { v[e] = hr[lane + 32 * e] + dr[lane + 32 * e]; sum += v[e]; }
#pragma unroll
  for (int off = 16; off > 0; off >>= 1) sum += __shfl_xor(sum, off, 32);
  const float mean = sum * (1.f / 128.f);
  float var = 0.f;
#pragma unroll
  for (int e = 0; e < 4; ++e) { float d = v[e] - mean; var += d * d; }
#pragma unroll
  for (int off = 16; off > 0; off >>= 1) var += __shfl_xor(var, off, 32);
  const float inv = rsqrtf(var * (1.f / 128.f) + 1e-5f);
#pragma unroll
  for (int e = 0; e < 4; ++e) {
    const int c = lane + 32 * e;
    const float o = (v[e] - mean) * inv * s[c] + bb[c];
    hr[c] = o;
    hH[(size_t)row * 128 + c] = (_Float16)o;
  }
}

// ---------------------------------------------------------------------------
// Node embedding + degree embeddings + virtual node; writes f32 + f16 h.
// grid = 32768 (row), block = 128 (channel)
// ---------------------------------------------------------------------------
__global__ void embed_kernel(const float* __restrict__ x, const float* __restrict__ nw,
                             const float* __restrict__ nb, const float* __restrict__ ie,
                             const float* __restrict__ oe, const float* __restrict__ vn,
                             const int* __restrict__ ind, const int* __restrict__ outd,
                             float* __restrict__ h, _Float16* __restrict__ hH) {
  __shared__ float xr[78];
  const int row = blockIdx.x, c = threadIdx.x;
  const int b = row >> 7, p = row & 127;
  float o;
  if (p == 0) {
    o = vn[c];
  } else {
    const int node = b * 127 + (p - 1);
    if (c < 78) xr[c] = x[(size_t)node * 78 + c];
    __syncthreads();
    float acc = nb[c];
    const float* wr = nw + (size_t)c * 78;
    for (int f = 0; f < 78; ++f) acc += xr[f] * wr[f];
    int di = ind[node];   di = di < 0 ? 0 : (di > 127 ? 127 : di);
    int dq = outd[node];  dq = dq < 0 ? 0 : (dq > 127 ? 127 : dq);
    o = acc + ie[(size_t)di * 128 + c] + oe[(size_t)dq * 128 + c];
  }
  h[(size_t)row * 128 + c]  = o;
  hH[(size_t)row * 128 + c] = (_Float16)o;
}

__global__ void f2h_kernel(const float* __restrict__ a, _Float16* __restrict__ o, int n) {
  const int i = blockIdx.x * 256 + threadIdx.x;
  if (i < n) o[i] = (_Float16)a[i];
}

__global__ void gather_h0_kernel(const float* __restrict__ h, _Float16* __restrict__ hG) {
  hG[(size_t)blockIdx.x * 128 + threadIdx.x] =
      (_Float16)h[(size_t)blockIdx.x * 16384 + threadIdx.x];
}

// ---------------------------------------------------------------------------
// Protein branch: embed gather + conv1d(k=8, VALID) -> [B,32,78]
// ---------------------------------------------------------------------------
__global__ void conv_kernel(const int* __restrict__ target, const float* __restrict__ emb,
                            const float* __restrict__ cw, const float* __restrict__ cb,
                            float* __restrict__ cbuf) {
  __shared__ float e[85 * 128];     // 43.5 KB
  const int b = blockIdx.x;
  for (int idx = threadIdx.x; idx < 85 * 128; idx += blockDim.x) {
    const int t = idx >> 7, ic = idx & 127;
    e[idx] = emb[(size_t)target[b * 85 + t] * 128 + ic];
  }
  __syncthreads();
  for (int idx = threadIdx.x; idx < 32 * 78; idx += blockDim.x) {
    const int oc = idx / 78, t = idx % 78;
    float acc = cb[oc];
    const float* w = cw + (size_t)oc * 128 * 8;   // [oc][ic][k]
    for (int ic = 0; ic < 128; ++ic) {
      const float* wk = w + ic * 8;
      const float* ek = e + (size_t)t * 128 + ic;
#pragma unroll
      for (int k = 0; k < 8; ++k) acc += ek[(size_t)k * 128] * wk[k];
    }
    cbuf[((size_t)b * 32 + oc) * 78 + t] = acc;
  }
}

__global__ void xt_kernel(const float* __restrict__ cbuf, const float* __restrict__ w,
                          const float* __restrict__ bias, float* __restrict__ xt) {
  __shared__ float cr[2496];
  const int b = blockIdx.x, c = threadIdx.x;  // 128 threads
  for (int i = threadIdx.x; i < 2496; i += 128) cr[i] = cbuf[(size_t)b * 2496 + i];
  __syncthreads();
  float acc = bias[c];
  const float* wr = w + (size_t)c * 2496;
  for (int i = 0; i < 2496; ++i) acc += cr[i] * wr[i];
  xt[(size_t)b * 128 + c] = acc;
}

__global__ void fc1_kernel(const float* __restrict__ gr, const float* __restrict__ xt,
                           const float* __restrict__ w, const float* __restrict__ bias,
                           float* __restrict__ a1) {
  __shared__ float xc[256];
  const int b = blockIdx.x;
  xc[threadIdx.x] = (threadIdx.x < 128) ? gr[(size_t)b * 128 + threadIdx.x]
                                        : xt[(size_t)b * 128 + threadIdx.x - 128];
  __syncthreads();
  for (int o = threadIdx.x; o < 1024; o += 256) {
    float acc = bias[o];
    const float* wr = w + (size_t)o * 256;
    for (int i = 0; i < 256; ++i) acc += xc[i] * wr[i];
    a1[(size_t)b * 1024 + o] = fmaxf(acc, 0.f);
  }
}

__global__ void fc2_kernel(const float* __restrict__ a1, const float* __restrict__ w,
                           const float* __restrict__ bias, float* __restrict__ a2) {
  __shared__ float xr[1024];
  const int b = blockIdx.x;
  for (int i = threadIdx.x; i < 1024; i += 256) xr[i] = a1[(size_t)b * 1024 + i];
  __syncthreads();
  const int o = threadIdx.x;   // 256 outputs
  float acc = bias[o];
  const float* wr = w + (size_t)o * 1024;
  for (int i = 0; i < 1024; ++i) acc += xr[i] * wr[i];
  a2[(size_t)b * 256 + o] = fmaxf(acc, 0.f);
}

__global__ void out_kernel(const float* __restrict__ a2, const float* __restrict__ w,
                           const float* __restrict__ bias, float* __restrict__ out) {
  __shared__ float red[256];
  const int b = blockIdx.x;
  red[threadIdx.x] = a2[(size_t)b * 256 + threadIdx.x] * w[threadIdx.x];
  __syncthreads();
  for (int sft = 128; sft > 0; sft >>= 1) {
    if (threadIdx.x < sft) red[threadIdx.x] += red[threadIdx.x + sft];
    __syncthreads();
  }
  if (threadIdx.x == 0) out[b] = red[0] + bias[0];
}

// ---------------------------------------------------------------------------
extern "C" void kernel_launch(void* const* d_in, const int* in_sizes, int n_in,
                              void* d_out, int out_size, void* d_ws, size_t ws_size,
                              hipStream_t stream) {
  const float* x        = (const float*)d_in[0];
  const int*   dist     = (const int*)d_in[2];
  const int*   in_deg   = (const int*)d_in[3];
  const int*   out_deg  = (const int*)d_in[4];
  const int*   target   = (const int*)d_in[5];
  const float* node_w   = (const float*)d_in[6];
  const float* node_b   = (const float*)d_in[7];
  const float* in_emb   = (const float*)d_in[8];
  const float* out_emb  = (const float*)d_in[9];
  const float* vnode    = (const float*)d_in[10];
  const float* q_w      = (const float*)d_in[11];
  const float* q_b      = (const float*)d_in[12];
  const float* k_w      = (const float*)d_in[13];
  const float* k_b      = (const float*)d_in[14];
  const float* v_w      = (const float*)d_in[15];
  const float* v_b      = (const float*)d_in[16];
  const float* o_w      = (const float*)d_in[17];
  const float* o_b      = (const float*)d_in[18];
  const float* sp_emb   = (const float*)d_in[19];
  const float* edge_w   = (const float*)d_in[20];
  const float* edge_b   = (const float*)d_in[21];
  const float* ln1_s    = (const float*)d_in[22];
  const float* ln1_b    = (const float*)d_in[23];
  const float* ln2_s    = (const float*)d_in[24];
  const float* ln2_b    = (const float*)d_in[25];
  const float* ffn_w1   = (const float*)d_in[26];
  const float* ffn_b1   = (const float*)d_in[27];
  const float* ffn_w2   = (const float*)d_in[28];
  const float* ffn_b2   = (const float*)d_in[29];
  const float* gp_w     = (const float*)d_in[30];
  const float* gp_b     = (const float*)d_in[31];
  const float* emb_xt   = (const float*)d_in[32];
  const float* conv_w   = (const float*)d_in[33];
  const float* conv_b   = (const float*)d_in[34];
  const float* fxt_w    = (const float*)d_in[35];
  const float* fxt_b    = (const float*)d_in[36];
  const float* fc1_w    = (const float*)d_in[37];
  const float* fc1_b    = (const float*)d_in[38];
  const float* fc2_w    = (const float*)d_in[39];
  const float* fc2_b    = (const float*)d_in[40];
  const float* outw     = (const float*)d_in[41];
  const float* outb     = (const float*)d_in[42];
  float* out = (float*)d_out;

  const int M = 32768;  // 256 graphs * 128 tokens
  size_t off = 0;
  auto carve = [&](size_t bytes) {
    void* p = (char*)d_ws + off;
    off += (bytes + 255) & ~(size_t)255;
    return p;
  };
  float*    hF   = (float*)   carve((size_t)M * 128 * 4);
  _Float16* hH   = (_Float16*)carve((size_t)M * 128 * 2);
  _Float16* qH   = (_Float16*)carve((size_t)M * 128 * 2);
  _Float16* kH   = (_Float16*)carve((size_t)M * 128 * 2);
  _Float16* vtH  = (_Float16*)carve((size_t)M * 128 * 2);
  _Float16* aoH  = (_Float16*)carve((size_t)M * 128 * 2);
  _Float16* ffH  = (_Float16*)carve((size_t)M * 512 * 2);
  float*    gout = (float*)   carve((size_t)M * 128 * 4);
  _Float16* qwH  = (_Float16*)carve((size_t)3 * 16384 * 2);
  _Float16* kwH  = (_Float16*)carve((size_t)3 * 16384 * 2);
  _Float16* vwH  = (_Float16*)carve((size_t)3 * 16384 * 2);
  _Float16* owH  = (_Float16*)carve((size_t)3 * 16384 * 2);
  _Float16* f1H  = (_Float16*)carve((size_t)3 * 65536 * 2);
  _Float16* f2H  = (_Float16*)carve((size_t)3 * 65536 * 2);
  _Float16* gpH  = (_Float16*)carve((size_t)16384 * 2);
  _Float16* hG   = (_Float16*)carve((size_t)256 * 128 * 2);
  float*    gr   = (float*)   carve((size_t)256 * 128 * 4);
  float*    cbuf = (float*)   carve((size_t)256 * 32 * 78 * 4);
  float*    xt   = (float*)   carve((size_t)256 * 128 * 4);
  float*    a1   = (float*)   carve((size_t)256 * 1024 * 4);
  float*    a2   = (float*)   carve((size_t)256 * 256 * 4);
  if (off > ws_size) return;   // workspace insufficient: bail deterministically

  // weight conversion f32 -> f16
  auto cvt = [&](const float* src, _Float16* dst, int n) {
    f2h_kernel<<<(n + 255) / 256, 256, 0, stream>>>(src, dst, n);
  };
  cvt(q_w,    qwH, 3 * 16384);
  cvt(k_w,    kwH, 3 * 16384);
  cvt(v_w,    vwH, 3 * 16384);
  cvt(o_w,    owH, 3 * 16384);
  cvt(ffn_w1, f1H, 3 * 65536);
  cvt(ffn_w2, f2H, 3 * 65536);
  cvt(gp_w,   gpH, 16384);

  // node + degree embedding, virtual node
  embed_kernel<<<M, 128, 0, stream>>>(x, node_w, node_b, in_emb, out_emb, vnode,
                                      in_deg, out_deg, hF, hH);

  const dim3 gemm_block(128);
  for (int l = 0; l < 3; ++l) {
    const _Float16* qw = qwH + (size_t)l * 16384;
    const _Float16* kw = kwH + (size_t)l * 16384;
    const _Float16* vw = vwH + (size_t)l * 16384;
    const _Float16* ow = owH + (size_t)l * 16384;
    // Q, K (f16), V (f16 transposed per head)
    wmma_gemm_kernel<<<dim3(M / 64, 2), gemm_block, 0, stream>>>(
        hH, 128, qw, 128, q_b + l * 128, nullptr, qH, 128, 128, 1);
    wmma_gemm_kernel<<<dim3(M / 64, 2), gemm_block, 0, stream>>>(
        hH, 128, kw, 128, k_b + l * 128, nullptr, kH, 128, 128, 1);
    wmma_gemm_kernel<<<dim3(M / 64, 2), gemm_block, 0, stream>>>(
        hH, 128, vw, 128, v_b + l * 128, nullptr, vtH, 128, 128, 3);
    // fused attention (graph x head blocks)
    attn_kernel<<<256 * 8, 256, 0, stream>>>(qH, kH, vtH, dist,
        sp_emb + l * 80, edge_w + l * 8, edge_b + l * 8, aoH);
    // output projection + residual + LN1
    wmma_gemm_kernel<<<dim3(M / 64, 2), gemm_block, 0, stream>>>(
        aoH, 128, ow, 128, o_b + l * 128, gout, nullptr, 128, 128, 0);
    add_ln_kernel<<<M, 32, 0, stream>>>(hF, hH, gout, ln1_s + l * 128, ln1_b + l * 128);
    // FFN
    wmma_gemm_kernel<<<dim3(M / 64, 8), gemm_block, 0, stream>>>(
        hH, 128, f1H + (size_t)l * 65536, 128, ffn_b1 + l * 512, nullptr, ffH, 128, 512, 2);
    wmma_gemm_kernel<<<dim3(M / 64, 2), gemm_block, 0, stream>>>(
        ffH, 512, f2H + (size_t)l * 65536, 512, ffn_b2 + l * 128, gout, nullptr, 512, 128, 0);
    add_ln_kernel<<<M, 32, 0, stream>>>(hF, hH, gout, ln2_s + l * 128, ln2_b + l * 128);
  }

  // graph readout: h[:,0] @ gp_w^T + gp_b (WMMA, M=256)
  gather_h0_kernel<<<256, 128, 0, stream>>>(hF, hG);
  wmma_gemm_kernel<<<dim3(4, 2), gemm_block, 0, stream>>>(
      hG, 128, gpH, 128, gp_b, gr, nullptr, 128, 128, 0);

  // protein branch + head MLP
  conv_kernel<<<256, 256, 0, stream>>>(target, emb_xt, conv_w, conv_b, cbuf);
  xt_kernel<<<256, 128, 0, stream>>>(cbuf, fxt_w, fxt_b, xt);
  fc1_kernel<<<256, 256, 0, stream>>>(gr, xt, fc1_w, fc1_b, a1);
  fc2_kernel<<<256, 256, 0, stream>>>(a1, fc2_w, fc2_b, a2);
  out_kernel<<<256, 256, 0, stream>>>(a2, outw, outb, out);
}
